// TabR_77799037600383
// MI455X (gfx1250) — matrix operand order repaired and checked by reading, not compiled, not run on hardware
//
#include <hip/hip_runtime.h>
#include <hip/hip_bf16.h>
#include <math.h>

#define D_IN    32
#define D_MAIN  128
#define D_BLOCK 256
#define CTX     96
#define LN_EPSF 1e-5f

typedef __attribute__((ext_vector_type(16))) __bf16 v16bf;
typedef __attribute__((ext_vector_type(8)))  float  v8f;

// ---------------------------------------------------------------------------
// Wave-local LDS fence: LDS ops are in-order per wave on CDNA5; the asm also
// acts as a compiler memory barrier for cross-lane LDS handoffs.
// ---------------------------------------------------------------------------
__device__ __forceinline__ void lds_fence() {
  asm volatile("s_wait_dscnt 0" ::: "memory");
}

// ---------------------------------------------------------------------------
// Fragment-native layouts (CDNA5 ISA 7.12.2, wave32):
//  A (16x32 bf16):  lane' = halfk*16 + m ; element j of lane' holds
//        k = (v>>2)*16 + halfk*8 + (v&3)*2 + p   (j = v*2+p)
//  B (32x16 bf16):  lane  = half*16 + n  ; element j holds k = (j>>3)*16 + half*8 + (j&7)
//  C/D (16x16 f32): VGPR r, lane half h, n = lane&15 -> C[r + 8h][n]
//
// Packed tiles: 32 lanes x 16 bf16 = 512 elements, lane-contiguous, so one
// fragment = one 32-byte vector load (global b128x2 / ds b128x2).
// ---------------------------------------------------------------------------
__device__ __forceinline__ int afrag_pos(int m, int k) {
  const int kt     = k >> 5;
  const int kk     = k & 31;
  const int grp    = kk >> 4;
  const int halfk  = (kk >> 3) & 1;
  const int within = (kk >> 1) & 3;
  const int p      = kk & 1;
  const int j      = ((((grp << 2) | within) << 1) | p);
  return kt * 512 + ((halfk << 4) + m) * 16 + j;
}

__device__ __forceinline__ v16bf a_frag(const v16bf* __restrict__ apack, int kt) {
  return apack[kt * 32 + (threadIdx.x & 31)];
}

__device__ __forceinline__ v16bf b_frag(const v16bf* __restrict__ Bp, int ktiles, int kt, int nt) {
  return Bp[(nt * ktiles + kt) * 32 + (threadIdx.x & 31)];
}

__device__ __forceinline__ v8f wmma_bf16(v16bf a, v16bf b, v8f c) {
  return __builtin_amdgcn_wmma_f32_16x16x32_bf16(false, a, false, b, (short)0, c, false, false);
}

// ---------------------------------------------------------------------------
// One-time weight repack: f32 row-major [K][N] -> fragment-native bf16 tiles.
// pos = ((nt*ktiles + kt)*32 + lane)*16 + j  maps to  W[kt*32 + (j>>3)*16 + (lane>>4)*8
// + (j&7)][nt*16 + (lane&15)].
// ---------------------------------------------------------------------------
__global__ __launch_bounds__(256)
void pack_kernel(const float* __restrict__ W, int K, int N, __bf16* __restrict__ out) {
  const int pos = blockIdx.x * 256 + threadIdx.x;
  if (pos >= K * N) return;
  const int j    = pos & 15;
  const int lane = (pos >> 4) & 31;
  const int tile = pos >> 9;
  const int ktiles = K >> 5;
  const int nt = tile / ktiles;
  const int kt = tile - nt * ktiles;
  const int k  = kt * 32 + ((j >> 3) << 4) + ((lane >> 4) << 3) + (j & 7);
  const int n  = nt * 16 + (lane & 15);
  out[pos] = (__bf16)W[k * N + n];
}

// ===========================================================================
// Encoder: per-wave, 16 rows through  lin -> block0 -> block1(prenorm) ->
// LN -> K projection.  Per-wave-private LDS, no block barriers.
// ===========================================================================
struct EncP {  // packed bf16 weights + f32 bias/norm params
  const v16bf *lin_Wp; const float *lin_b;
  const v16bf *e0_W1p; const float *e0_b1; const v16bf *e0_W2p; const float *e0_b2;
  const float *e1_g, *e1_b;
  const v16bf *e1_W1p; const float *e1_b1; const v16bf *e1_W2p; const float *e1_b2;
  const float *mix_g, *mix_b;
  const v16bf *K_Wp; const float *K_b;
};

#define ENC_WAVES 2

__device__ void mlp_apply(float (*hbuf)[D_MAIN], v16bf* apack,
                          const v16bf* __restrict__ W1p, const float* __restrict__ b1,
                          const v16bf* __restrict__ W2p, const float* __restrict__ b2) {
  const int lane = threadIdx.x & 31;
  const int half = lane >> 4;
  const int ncol = lane & 15;
  __bf16* ap = (__bf16*)apack;

  v16bf a[4];
#pragma unroll
  for (int kk = 0; kk < 4; ++kk) a[kk] = a_frag(apack, kk);

  // u = relu(n @ W1 + b1) scattered into A-fragment order (K = 256 tiles)
  for (int t = 0; t < 16; ++t) {
    v8f acc = {};
#pragma unroll
    for (int kk = 0; kk < 4; ++kk)
      acc = wmma_bf16(a[kk], b_frag(W1p, 4, kk, t), acc);
    const int col = t * 16 + ncol;
    const float bias = b1[col];
#pragma unroll
    for (int r = 0; r < 8; ++r) {
      const float u = acc[r] + bias;
      ap[afrag_pos(r + 8 * half, col)] = (__bf16)(u > 0.f ? u : 0.f);
    }
  }
  lds_fence();

  // h += u @ W2 + b2
  v16bf a2[8];
#pragma unroll
  for (int kk = 0; kk < 8; ++kk) a2[kk] = a_frag(apack, kk);
  for (int t = 0; t < 8; ++t) {
    v8f acc = {};
#pragma unroll
    for (int kk = 0; kk < 8; ++kk)
      acc = wmma_bf16(a2[kk], b_frag(W2p, 8, kk, t), acc);
    const int col = t * 16 + ncol;
    const float bias = b2[col];
#pragma unroll
    for (int r = 0; r < 8; ++r)
      hbuf[r + 8 * half][col] += acc[r] + bias;
  }
  lds_fence();
}

__device__ void copy_to_apack(float (*hbuf)[D_MAIN], v16bf* apack) {
  const int lane = threadIdx.x & 31;
  __bf16* ap = (__bf16*)apack;
  for (int e = lane; e < 16 * D_MAIN; e += 32) {
    const int r = e >> 7, c = e & 127;
    ap[afrag_pos(r, c)] = (__bf16)hbuf[r][c];
  }
  lds_fence();
}

__device__ void ln_to_apack(float (*hbuf)[D_MAIN], v16bf* apack,
                            const float* __restrict__ g, const float* __restrict__ b) {
  const int lane = threadIdx.x & 31;
  const int half = lane >> 4;
  const int m    = lane & 15;
  __bf16* ap = (__bf16*)apack;
  float s = 0.f, ss = 0.f;
  for (int c = 0; c < D_MAIN; ++c) { const float v = hbuf[m][c]; s += v; ss += v * v; }
  const float mu   = s * (1.f / D_MAIN);
  const float var  = ss * (1.f / D_MAIN) - mu * mu;
  const float rstd = rsqrtf(var + LN_EPSF);
  const int c0 = half * 64;
  for (int c = c0; c < c0 + 64; ++c)
    ap[afrag_pos(m, c)] = (__bf16)((hbuf[m][c] - mu) * rstd * g[c] + b[c]);
  lds_fence();
}

__global__ __launch_bounds__(ENC_WAVES * 32)
void encode_kernel(const float* __restrict__ X, int nrows, EncP w,
                   float* __restrict__ k_out, float* __restrict__ ksq_out,
                   float* __restrict__ h_out, int write_h) {
  __shared__ float hbuf_s[ENC_WAVES][16][D_MAIN];
  __shared__ v16bf apack_s[ENC_WAVES][8 * 32];      // 8 K-tiles (K up to 256)
  __shared__ float kf_s[ENC_WAVES][16][D_MAIN];

  const int wave = threadIdx.x >> 5;
  const int lane = threadIdx.x & 31;
  const int half = lane >> 4;
  const int ncol = lane & 15;
  const int row_base = (blockIdx.x * ENC_WAVES + wave) * 16;
  if (row_base >= nrows) return;

  float (*hbuf)[D_MAIN] = hbuf_s[wave];
  v16bf* apack          = apack_s[wave];
  float (*kf)[D_MAIN]   = kf_s[wave];
  __bf16* ap = (__bf16*)apack;

  // 1) stage input rows (16 x 32) directly in A-fragment order
  for (int e = lane; e < 16 * D_IN; e += 32) {
    const int r = e >> 5, c = e & 31;
    const int gr = row_base + r;
    ap[afrag_pos(r, c)] = (__bf16)((gr < nrows) ? X[(size_t)gr * D_IN + c] : 0.f);
  }
  lds_fence();

  // 2) h = x @ lin_W + lin_b   (K = 32)
  {
    v16bf a = a_frag(apack, 0);
    for (int t = 0; t < 8; ++t) {
      v8f acc = {};
      acc = wmma_bf16(a, b_frag(w.lin_Wp, 1, 0, t), acc);
      const int col = t * 16 + ncol;
      const float bias = w.lin_b[col];
#pragma unroll
      for (int r = 0; r < 8; ++r)
        hbuf[r + 8 * half][col] = acc[r] + bias;
    }
  }
  lds_fence();

  // 3) block0 (no prenorm), 4) block1 (prenorm)
  copy_to_apack(hbuf, apack);
  mlp_apply(hbuf, apack, w.e0_W1p, w.e0_b1, w.e0_W2p, w.e0_b2);
  ln_to_apack(hbuf, apack, w.e1_g, w.e1_b);
  mlp_apply(hbuf, apack, w.e1_W1p, w.e1_b1, w.e1_W2p, w.e1_b2);

  // 5) optional h output (query batch only)
  if (write_h) {
    for (int e = lane; e < 16 * D_MAIN; e += 32) {
      const int r = e >> 7, c = e & 127;
      const int gr = row_base + r;
      if (gr < nrows) h_out[(size_t)gr * D_MAIN + c] = hbuf[r][c];
    }
  }

  // 6) k = LN(h) @ K_W + K_b ; also ||k||^2
  ln_to_apack(hbuf, apack, w.mix_g, w.mix_b);
  {
    v16bf a[4];
#pragma unroll
    for (int kk = 0; kk < 4; ++kk) a[kk] = a_frag(apack, kk);
    for (int t = 0; t < 8; ++t) {
      v8f acc = {};
#pragma unroll
      for (int kk = 0; kk < 4; ++kk)
        acc = wmma_bf16(a[kk], b_frag(w.K_Wp, 4, kk, t), acc);
      const int col = t * 16 + ncol;
      const float bias = w.K_b[col];
#pragma unroll
      for (int r = 0; r < 8; ++r) {
        const int m = r + 8 * half;
        const float kv = acc[r] + bias;
        kf[m][col] = kv;
        const int gr = row_base + m;
        if (gr < nrows) k_out[(size_t)gr * D_MAIN + col] = kv;
      }
    }
  }
  lds_fence();
  {
    const int m = lane & 15;
    float s = 0.f;
    for (int c = half * 64; c < half * 64 + 64; ++c) {
      const float v = kf[m][c];
      s += v * v;
    }
    s += __shfl_xor(s, 16, 32);
    if (half == 0 && row_base + m < nrows) ksq_out[row_base + m] = s;
  }
}

// ===========================================================================
// kNN selection: one block per query.  key = ||c||^2 - 2*q.c  (||q||^2
// cancels in ranking AND softmax).  Deterministic threshold pool + bitonic
// sort by (key, idx).
// ===========================================================================
#define POOL        4096
#define SEL_THREADS 256
#define SEL_CHUNK   2048

__device__ void pool_sort(float* pk, int* pi, int cnt_val) {
  const int tid = threadIdx.x;
  for (int e = tid; e < POOL; e += SEL_THREADS)
    if (e >= cnt_val) { pk[e] = __builtin_inff(); pi[e] = 0x7fffffff; }
  __syncthreads();
  for (int k = 2; k <= POOL; k <<= 1) {
    for (int j = k >> 1; j > 0; j >>= 1) {
      for (int i = tid; i < POOL; i += SEL_THREADS) {
        const int l = i ^ j;
        if (l > i) {
          const bool up = ((i & k) == 0);
          const float k1 = pk[i], k2 = pk[l];
          const int   i1 = pi[i], i2 = pi[l];
          const bool gt = (k1 > k2) || (k1 == k2 && i1 > i2);
          if (gt == up) { pk[i] = k2; pk[l] = k1; pi[i] = i2; pi[l] = i1; }
        }
      }
      __syncthreads();
    }
  }
}

__global__ __launch_bounds__(SEL_THREADS)
void select_kernel(const float* __restrict__ xk, const float* __restrict__ cand_k,
                   const float* __restrict__ cand_ksq, int ncand,
                   int* __restrict__ ctx_idx, float* __restrict__ probs) {
  __shared__ float kq[D_MAIN];
  __shared__ float pk[POOL];
  __shared__ int   pi[POOL];
  __shared__ int   cnt_s;
  __shared__ float tau_s;
  __shared__ float earr[CTX];
  __shared__ float red[128];

  const int q = blockIdx.x;
  const int tid = threadIdx.x;
  if (tid < D_MAIN) kq[tid] = xk[(size_t)q * D_MAIN + tid];
  if (tid == 0) { cnt_s = 0; tau_s = __builtin_inff(); }
  __syncthreads();

  for (int base = 0; base < ncand; base += SEL_CHUNK) {
    const float tcur = tau_s;
#pragma unroll 1
    for (int i = 0; i < SEL_CHUNK / SEL_THREADS; ++i) {
      const int n = base + i * SEL_THREADS + tid;
      if (n < ncand) {
        const float4* row = (const float4*)(cand_k + (size_t)n * D_MAIN);
        const float4* q4  = (const float4*)kq;
        float dot = 0.f;
#pragma unroll
        for (int c4 = 0; c4 < D_MAIN / 4; ++c4) {
          const float4 a = row[c4], b = q4[c4];
          dot += a.x * b.x + a.y * b.y + a.z * b.z + a.w * b.w;
        }
        const float key = cand_ksq[n] - 2.f * dot;
        if (key < tcur) {
          const int p = atomicAdd(&cnt_s, 1);
          if (p < POOL) { pk[p] = key; pi[p] = n; }
        }
      }
    }
    __syncthreads();
    if (cnt_s > SEL_CHUNK) {       // uniform (shared) condition
      const int cv = cnt_s;
      pool_sort(pk, pi, cv);
      if (tid == 0) { tau_s = pk[CTX - 1]; cnt_s = (cv < CTX) ? cv : CTX; }
    }
    __syncthreads();
  }

  const int cfin = cnt_s;
  pool_sort(pk, pi, cfin);
  if (tid < CTX) {
    const float e = expf(pk[0] - pk[tid]);
    earr[tid] = e;
    red[tid]  = e;
  } else if (tid < 128) {
    red[tid] = 0.f;
  }
  __syncthreads();
  for (int s2 = 64; s2 > 0; s2 >>= 1) {
    if (tid < s2) red[tid] += red[tid + s2];
    __syncthreads();
  }
  if (tid < CTX) {
    probs[(size_t)q * CTX + tid]   = earr[tid] / red[0];
    ctx_idx[(size_t)q * CTX + tid] = pi[tid];
  }
}

// ===========================================================================
// T-module + value aggregation: one wave per query, deterministic register
// accumulation of the probs-weighted sum (cross-half combine via shfl_xor 16).
// ===========================================================================
#define TW 2

__global__ __launch_bounds__(TW * 32)
void tmod_kernel(const float* __restrict__ xk, const float* __restrict__ xh,
                 const float* __restrict__ cand_k, const float* __restrict__ cand_y,
                 const int* __restrict__ ctx_idx, const float* __restrict__ probs,
                 const v16bf* __restrict__ T_W1p, const float* __restrict__ T_b1,
                 const v16bf* __restrict__ T_W2p,
                 const float* __restrict__ lab_w, const float* __restrict__ lab_b,
                 float* __restrict__ x2, int nq) {
  __shared__ float kq_s[TW][D_MAIN];
  __shared__ v16bf apack_s[TW][8 * 32];
  __shared__ int   jidx_s[TW][16];
  __shared__ float yv_s[TW][16];
  __shared__ float pv_s[TW][16];

  const int wave = threadIdx.x >> 5;
  const int lane = threadIdx.x & 31;
  const int half = lane >> 4;
  const int ncol = lane & 15;
  const int q = blockIdx.x * TW + wave;
  if (q >= nq) return;

  float* kq    = kq_s[wave];
  v16bf* apack = apack_s[wave];
  int*   jidx  = jidx_s[wave];
  float* yv    = yv_s[wave];
  float* pv    = pv_s[wave];
  __bf16* ap = (__bf16*)apack;

  for (int c = lane; c < D_MAIN; c += 32) kq[c] = xk[(size_t)q * D_MAIN + c];

  float acc[8][8];
#pragma unroll
  for (int t = 0; t < 8; ++t)
#pragma unroll
    for (int r = 0; r < 8; ++r) acc[t][r] = 0.f;

  for (int jt = 0; jt < CTX / 16; ++jt) {
    if (lane < 16) {
      const int j = jt * 16 + lane;
      const int idx = ctx_idx[(size_t)q * CTX + j];
      jidx[lane] = idx;
      yv[lane]   = cand_y[idx];
      pv[lane]   = probs[(size_t)q * CTX + j];
    }
    lds_fence();

    // diff = k_q - context_k, scattered in A-fragment order (K=128 tiles 0..3)
    for (int e = lane; e < 16 * D_MAIN; e += 32) {
      const int r = e >> 7, c = e & 127;
      ap[afrag_pos(r, c)] = (__bf16)(kq[c] - cand_k[(size_t)jidx[r] * D_MAIN + c]);
    }
    lds_fence();

    // u = relu(diff @ T_W1 + T_b1), scattered back (K=256 tiles 0..7)
    v16bf a[4];
#pragma unroll
    for (int kk = 0; kk < 4; ++kk) a[kk] = a_frag(apack, kk);
    for (int t = 0; t < 16; ++t) {
      v8f c8 = {};
#pragma unroll
      for (int kk = 0; kk < 4; ++kk)
        c8 = wmma_bf16(a[kk], b_frag(T_W1p, 4, kk, t), c8);
      const int col = t * 16 + ncol;
      const float bias = T_b1[col];
#pragma unroll
      for (int r = 0; r < 8; ++r) {
        const float u = c8[r] + bias;
        ap[afrag_pos(r + 8 * half, col)] = (__bf16)(u > 0.f ? u : 0.f);
      }
    }
    lds_fence();

    // t = u @ T_W2 ; value = y*lab_w + lab_b + t ; acc += p * value
    v16bf a2[8];
#pragma unroll
    for (int kk = 0; kk < 8; ++kk) a2[kk] = a_frag(apack, kk);
    for (int t = 0; t < 8; ++t) {
      v8f c8 = {};
#pragma unroll
      for (int kk = 0; kk < 8; ++kk)
        c8 = wmma_bf16(a2[kk], b_frag(T_W2p, 8, kk, t), c8);
      const int col = t * 16 + ncol;
      const float lw = lab_w[col], lb = lab_b[col];
#pragma unroll
      for (int r = 0; r < 8; ++r) {
        const int m = r + 8 * half;
        const float val = c8[r] + yv[m] * lw + lb;
        acc[t][r] += pv[m] * val;
      }
    }
    lds_fence();
  }

#pragma unroll
  for (int t = 0; t < 8; ++t) {
    float s = 0.f;
#pragma unroll
    for (int r = 0; r < 8; ++r) s += acc[t][r];
    s += __shfl_xor(s, 16, 32);
    if (half == 0) {
      const int col = t * 16 + ncol;
      x2[(size_t)q * D_MAIN + col] = xh[(size_t)q * D_MAIN + col] + s;
    }
  }
}

// ===========================================================================
// Predictor block (prenorm MLP) + head.  One block (128 threads) per row.
// ===========================================================================
__global__ __launch_bounds__(128)
void head_kernel(const float* __restrict__ x2,
                 const float* __restrict__ p0_g, const float* __restrict__ p0_b,
                 const float* __restrict__ p0_W1, const float* __restrict__ p0_b1,
                 const float* __restrict__ p0_W2, const float* __restrict__ p0_b2,
                 const float* __restrict__ head_g, const float* __restrict__ head_b,
                 const float* __restrict__ head_W, const float* __restrict__ head_bias,
                 float* __restrict__ out) {
  __shared__ float xr[D_MAIN];
  __shared__ float nr[D_MAIN];
  __shared__ float ub[D_BLOCK];
  __shared__ float red[128];

  const int row = blockIdx.x;
  const int tid = threadIdx.x;
  xr[tid] = x2[(size_t)row * D_MAIN + tid];
  __syncthreads();

  float s = 0.f, ss = 0.f;
  for (int c = 0; c < D_MAIN; ++c) { const float v = xr[c]; s += v; ss += v * v; }
  float mu = s / D_MAIN, var = ss / D_MAIN - mu * mu, rstd = rsqrtf(var + LN_EPSF);
  nr[tid] = (xr[tid] - mu) * rstd * p0_g[tid] + p0_b[tid];
  __syncthreads();

  for (int t = 0; t < 2; ++t) {
    const int col = tid + t * 128;
    float a = p0_b1[col];
    for (int c = 0; c < D_MAIN; ++c) a += nr[c] * p0_W1[c * D_BLOCK + col];
    ub[col] = a > 0.f ? a : 0.f;
  }
  __syncthreads();

  float a = p0_b2[tid];
  for (int c = 0; c < D_BLOCK; ++c) a += ub[c] * p0_W2[c * D_MAIN + tid];
  const float x3 = xr[tid] + a;
  __syncthreads();
  xr[tid] = x3;
  __syncthreads();

  s = 0.f; ss = 0.f;
  for (int c = 0; c < D_MAIN; ++c) { const float v = xr[c]; s += v; ss += v * v; }
  mu = s / D_MAIN; var = ss / D_MAIN - mu * mu; rstd = rsqrtf(var + LN_EPSF);
  float h = (xr[tid] - mu) * rstd * head_g[tid] + head_b[tid];
  h = h > 0.f ? h : 0.f;
  red[tid] = h * head_W[tid];
  __syncthreads();
  for (int s2 = 64; s2 > 0; s2 >>= 1) {
    if (tid < s2) red[tid] += red[tid + s2];
    __syncthreads();
  }
  if (tid == 0) out[row] = red[0] + head_bias[0];
}

// ===========================================================================
// Launch
// ===========================================================================
extern "C" void kernel_launch(void* const* d_in, const int* in_sizes, int n_in,
                              void* d_out, int out_size, void* d_ws, size_t ws_size,
                              hipStream_t stream) {
  const float* x_num  = (const float*)d_in[0];
  const float* cand_x = (const float*)d_in[1];
  const float* cand_y = (const float*)d_in[2];
  // d_in[3] = context_size (== 96, compile-time constant here)

  const float* lin_W = (const float*)d_in[4];  const float* lin_b = (const float*)d_in[5];
  const float* e0_W1 = (const float*)d_in[6];  const float* e0_b1 = (const float*)d_in[7];
  const float* e0_W2 = (const float*)d_in[8];  const float* e0_b2 = (const float*)d_in[9];
  const float* e1_g  = (const float*)d_in[10]; const float* e1_b  = (const float*)d_in[11];
  const float* e1_W1 = (const float*)d_in[12]; const float* e1_b1 = (const float*)d_in[13];
  const float* e1_W2 = (const float*)d_in[14]; const float* e1_b2 = (const float*)d_in[15];
  const float* mix_g = (const float*)d_in[16]; const float* mix_b = (const float*)d_in[17];
  const float* K_W   = (const float*)d_in[18]; const float* K_b   = (const float*)d_in[19];
  const float* T_W1  = (const float*)d_in[20];
  const float* T_b1  = (const float*)d_in[21];
  const float* T_W2  = (const float*)d_in[22];
  const float* lab_w = (const float*)d_in[23];
  const float* lab_b = (const float*)d_in[24];
  const float* p0_g  = (const float*)d_in[25];
  const float* p0_b  = (const float*)d_in[26];
  const float* p0_W1 = (const float*)d_in[27];
  const float* p0_b1 = (const float*)d_in[28];
  const float* p0_W2 = (const float*)d_in[29];
  const float* p0_b2 = (const float*)d_in[30];
  const float* head_g    = (const float*)d_in[31];
  const float* head_b    = (const float*)d_in[32];
  const float* head_W    = (const float*)d_in[33];
  const float* head_bias = (const float*)d_in[34];

  const int NC = in_sizes[2];          // 100000 candidates
  const int NQ = in_sizes[0] / D_IN;   // 256 query rows

  // workspace layout (~52.6 MB)
  char* ws = (char*)d_ws;
  size_t off = 0;
  auto wsalloc = [&](size_t bytes) -> void* {
    void* p = (void*)(ws + off);
    off += (bytes + 255) & ~(size_t)255;
    return p;
  };
  float* cand_k   = (float*)wsalloc((size_t)NC * D_MAIN * sizeof(float));
  float* cand_ksq = (float*)wsalloc((size_t)NC * sizeof(float));
  float* xk       = (float*)wsalloc((size_t)NQ * D_MAIN * sizeof(float));
  float* xh       = (float*)wsalloc((size_t)NQ * D_MAIN * sizeof(float));
  float* xksq     = (float*)wsalloc((size_t)NQ * sizeof(float));
  int*   ctx_idx  = (int*)  wsalloc((size_t)NQ * CTX * sizeof(int));
  float* probs    = (float*)wsalloc((size_t)NQ * CTX * sizeof(float));
  float* x2       = (float*)wsalloc((size_t)NQ * D_MAIN * sizeof(float));
  // packed bf16 weights (fragment-native tiles)
  __bf16* lin_Wp = (__bf16*)wsalloc((size_t)D_IN    * D_MAIN  * 2);
  __bf16* e0_W1p = (__bf16*)wsalloc((size_t)D_MAIN  * D_BLOCK * 2);
  __bf16* e0_W2p = (__bf16*)wsalloc((size_t)D_BLOCK * D_MAIN  * 2);
  __bf16* e1_W1p = (__bf16*)wsalloc((size_t)D_MAIN  * D_BLOCK * 2);
  __bf16* e1_W2p = (__bf16*)wsalloc((size_t)D_BLOCK * D_MAIN  * 2);
  __bf16* K_Wp   = (__bf16*)wsalloc((size_t)D_MAIN  * D_MAIN  * 2);
  __bf16* T_W1p  = (__bf16*)wsalloc((size_t)D_MAIN  * D_BLOCK * 2);
  __bf16* T_W2p  = (__bf16*)wsalloc((size_t)D_BLOCK * D_MAIN  * 2);
  (void)ws_size; (void)n_in; (void)out_size;

  // 0) repack weights into fragment-native bf16 tiles
  pack_kernel<<<(D_IN    * D_MAIN ) / 256, 256, 0, stream>>>(lin_W, D_IN,    D_MAIN,  lin_Wp);
  pack_kernel<<<(D_MAIN  * D_BLOCK) / 256, 256, 0, stream>>>(e0_W1, D_MAIN,  D_BLOCK, e0_W1p);
  pack_kernel<<<(D_BLOCK * D_MAIN ) / 256, 256, 0, stream>>>(e0_W2, D_BLOCK, D_MAIN,  e0_W2p);
  pack_kernel<<<(D_MAIN  * D_BLOCK) / 256, 256, 0, stream>>>(e1_W1, D_MAIN,  D_BLOCK, e1_W1p);
  pack_kernel<<<(D_BLOCK * D_MAIN ) / 256, 256, 0, stream>>>(e1_W2, D_BLOCK, D_MAIN,  e1_W2p);
  pack_kernel<<<(D_MAIN  * D_MAIN ) / 256, 256, 0, stream>>>(K_W,   D_MAIN,  D_MAIN,  K_Wp);
  pack_kernel<<<(D_MAIN  * D_BLOCK) / 256, 256, 0, stream>>>(T_W1,  D_MAIN,  D_BLOCK, T_W1p);
  pack_kernel<<<(D_BLOCK * D_MAIN ) / 256, 256, 0, stream>>>(T_W2,  D_BLOCK, D_MAIN,  T_W2p);

  EncP w;
  w.lin_Wp = (const v16bf*)lin_Wp; w.lin_b = lin_b;
  w.e0_W1p = (const v16bf*)e0_W1p; w.e0_b1 = e0_b1;
  w.e0_W2p = (const v16bf*)e0_W2p; w.e0_b2 = e0_b2;
  w.e1_g = e1_g; w.e1_b = e1_b;
  w.e1_W1p = (const v16bf*)e1_W1p; w.e1_b1 = e1_b1;
  w.e1_W2p = (const v16bf*)e1_W2p; w.e1_b2 = e1_b2;
  w.mix_g = mix_g; w.mix_b = mix_b;
  w.K_Wp = (const v16bf*)K_Wp; w.K_b = K_b;

  const int rows_per_blk = ENC_WAVES * 16;

  // 1) encode candidates -> cand_k, cand_ksq
  encode_kernel<<<(NC + rows_per_blk - 1) / rows_per_blk, ENC_WAVES * 32, 0, stream>>>(
      cand_x, NC, w, cand_k, cand_ksq, x2 /*unused*/, 0);

  // 2) encode queries -> xk, xh
  encode_kernel<<<(NQ + rows_per_blk - 1) / rows_per_blk, ENC_WAVES * 32, 0, stream>>>(
      x_num, NQ, w, xk, xksq, xh, 1);

  // 3) kNN + softmax probs
  select_kernel<<<NQ, SEL_THREADS, 0, stream>>>(xk, cand_k, cand_ksq, NC, ctx_idx, probs);

  // 4) T-module + weighted aggregation -> x2
  tmod_kernel<<<(NQ + TW - 1) / TW, TW * 32, 0, stream>>>(
      xk, xh, cand_k, cand_y, ctx_idx, probs,
      (const v16bf*)T_W1p, T_b1, (const v16bf*)T_W2p, lab_w, lab_b, x2, NQ);

  // 5) predictor block + head -> out [NQ]
  head_kernel<<<NQ, 128, 0, stream>>>(x2, p0_g, p0_b, p0_W1, p0_b1, p0_W2, p0_b2,
                                      head_g, head_b, head_W, head_bias, (float*)d_out);
}
